// Qwen3MoeAttention_43885975830867
// MI455X (gfx1250) — compile-verified, matrix operand order checked by reading, not run on hardware
//
#include <hip/hip_runtime.h>

#define NB 2
#define NT 2048
#define NC 2048
#define NH 16
#define NKVH 4
#define ND 128

typedef __attribute__((ext_vector_type(16))) __bf16       v16bf;
typedef __attribute__((ext_vector_type(8)))  float        v8f;
typedef __attribute__((ext_vector_type(8)))  unsigned int v8u;
typedef unsigned int u32x4 __attribute__((ext_vector_type(4)));
typedef int          i32x8 __attribute__((ext_vector_type(8)));
typedef int          i32x4 __attribute__((ext_vector_type(4)));

__device__ __forceinline__ unsigned short f2bf(float f) {
  unsigned int u = __float_as_uint(f);
  u += 0x7fffu + ((u >> 16) & 1u);          // round to nearest even
  return (unsigned short)(u >> 16);
}
__device__ __forceinline__ float bf2f(unsigned short s) {
  return __uint_as_float(((unsigned int)s) << 16);
}
__device__ __forceinline__ unsigned int pk2bf(float a, float b) {
  return (unsigned int)f2bf(a) | ((unsigned int)f2bf(b) << 16);
}
__device__ __forceinline__ v8f vzero() {
  v8f z;
#pragma unroll
  for (int i = 0; i < 8; ++i) z[i] = 0.0f;
  return z;
}

// Build a bf16 WMMA A/B fragment from one LDS row (row-major along K, 32 K values).
// ISA 7.12.2: lanes 0-15 hold K={0..7,16..23}, lanes 16-31 hold K={8..15,24..31},
// two consecutive K per VGPR -> aligned 32-bit pair loads (lowers to ds_load_b128 x2).
__device__ __forceinline__ v16bf make_frag(const unsigned short* rowptr, int half) {
  const unsigned int* p = (const unsigned int*)rowptr;
  v8u u;
#pragma unroll
  for (int q = 0; q < 8; ++q) {
    int kbase = 2 * q + 8 * half + ((q >= 4) ? 8 : 0);
    u[q] = p[kbase >> 1];
  }
  return __builtin_bit_cast(v16bf, u);
}

__device__ __forceinline__ float rmax16(float v) {
#pragma unroll
  for (int m = 1; m <= 8; m <<= 1) v = fmaxf(v, __shfl_xor(v, m, 32));
  return v;
}
__device__ __forceinline__ float rsum16(float v) {
#pragma unroll
  for (int m = 1; m <= 8; m <<= 1) v += __shfl_xor(v, m, 32);
  return v;
}

// ---------------------------------------------------------------------------
// TDM: DMA a 2D bf16 tile (tile_rows x tile_cols, row stride stride_elems)
// from global memory into LDS at lds_byte_addr.  D# per ISA ch.8:
//  group0: count=1 | lds_addr | global_addr[56:0] | type=2
//  group1: data_size=2B, tensor_dim0/1, tile_dim0/1, tensor_dim0_stride
// All operands are wave-uniform (SGPR-resident).  Tracked by TENSORcnt,
// completes in issue order -> non-zero s_wait_tensorcnt enables pipelining.
// ---------------------------------------------------------------------------
__device__ __forceinline__ void tdm_load_2d(unsigned lds_byte_addr,
                                            const unsigned short* gptr,
                                            int tile_cols, int tile_rows,
                                            int stride_elems, int tensor_rows) {
  unsigned long long ga = (unsigned long long)(size_t)gptr;
  u32x4 g0;
  g0[0] = 1u;                                            // count=1, user mode
  g0[1] = lds_byte_addr;                                 // bits 63:32
  g0[2] = (unsigned)(ga & 0xffffffffu);                  // global_addr lo
  g0[3] = (unsigned)((ga >> 32) & 0x01ffffffu) | (2u << 30);  // addr hi | type=2
  i32x8 g1;
  g1[0] = 0x00010000;                                    // data_size = 1 (2 bytes)
  g1[1] = (int)((stride_elems & 0xffff) << 16);          // tensor_dim0 lo16
  g1[2] = (int)(((unsigned)stride_elems >> 16) & 0xffffu) |
          (int)((tensor_rows & 0xffff) << 16);           // dim0 hi | tensor_dim1 lo
  g1[3] = (int)(((unsigned)tensor_rows >> 16) & 0xffffu) |
          (int)((tile_cols & 0xffff) << 16);             // dim1 hi | tile_dim0
  g1[4] = tile_rows & 0xffff;                            // tile_dim1 (tile_dim2=0)
  g1[5] = stride_elems;                                  // tensor_dim0_stride lo32
  g1[6] = 0;
  g1[7] = 0;
  i32x4 z4 = {0, 0, 0, 0};                               // 2D: groups 2/3 null
  i32x8 z8 = {0, 0, 0, 0, 0, 0, 0, 0};
  __builtin_amdgcn_tensor_load_to_lds(g0, g1, z4, z4, z8, 0);
}
__device__ __forceinline__ unsigned lds_addr_of(const void* p) {
  return (unsigned)(size_t)p;                            // low 32 bits = LDS offset
}

// ---------------------------------------------------------------------------
// Kernel 0: one-time fp32 -> bf16 conversion (4 elems/thread)
// ---------------------------------------------------------------------------
__global__ __launch_bounds__(256) void cvt_bf16(const float* __restrict__ src,
                                                unsigned short* __restrict__ dst,
                                                int n4) {
  int i = blockIdx.x * 256 + threadIdx.x;
  if (i < n4) {
    float4 f = ((const float4*)src)[i];
    uint2 o;
    o.x = pk2bf(f.x, f.y);
    o.y = pk2bf(f.z, f.w);
    ((uint2*)dst)[i] = o;
  }
}

// ---------------------------------------------------------------------------
// Kernel 1: Y = X * W^T + b  (X: M x K bf16, W: N x K bf16), bf16 output
// scattered into (b, head, t, d) layout.  64x64 block tile, 4 waves.
// Double-buffered TDM: tiles for iteration i+1 stream into the alternate
// LDS buffers while WMMAs consume iteration i (s_wait_tensorcnt 2).
// ---------------------------------------------------------------------------
__global__ __launch_bounds__(128) void gemm_qkv(
    const unsigned short* __restrict__ X, const unsigned short* __restrict__ W,
    const float* __restrict__ bias, unsigned short* __restrict__ outb,
    int Kdim, int Mdim, int Ndim, int nheads)
{
  __shared__ __attribute__((aligned(16))) unsigned short As[2][64][32];
  __shared__ __attribute__((aligned(16))) unsigned short Bs[2][64][32];
  const int tid = threadIdx.x;
  const int lane = tid & 31, wave = tid >> 5;
  const int half = lane >> 4, l16 = lane & 15;
  const int m0 = blockIdx.x * 64;
  const int n0 = blockIdx.y * 64;
  unsigned ldsA[2], ldsB[2];
  ldsA[0] = lds_addr_of(&As[0][0][0]); ldsA[1] = lds_addr_of(&As[1][0][0]);
  ldsB[0] = lds_addr_of(&Bs[0][0][0]); ldsB[1] = lds_addr_of(&Bs[1][0][0]);

  v8f acc[4];
#pragma unroll
  for (int t = 0; t < 4; ++t) acc[t] = vzero();

  const int nk = Kdim / 32;
  if (wave == 0) {                                       // prime buffer 0
    tdm_load_2d(ldsA[0], X + (size_t)m0 * Kdim, 32, 64, Kdim, Mdim);
    tdm_load_2d(ldsB[0], W + (size_t)n0 * Kdim, 32, 64, Kdim, Ndim);
  }
  for (int ki = 0; ki < nk; ++ki) {
    const int cur = ki & 1, nxt = cur ^ 1;
    if (wave == 0) {
      if (ki + 1 < nk) {                                 // prefetch next tiles
        const int k1 = (ki + 1) * 32;
        tdm_load_2d(ldsA[nxt], X + (size_t)m0 * Kdim + k1, 32, 64, Kdim, Mdim);
        tdm_load_2d(ldsB[nxt], W + (size_t)n0 * Kdim + k1, 32, 64, Kdim, Ndim);
        __builtin_amdgcn_s_wait_tensorcnt(2);            // cur pair landed
      } else {
        __builtin_amdgcn_s_wait_tensorcnt(0);
      }
    }
    __syncthreads();
    v16bf af = make_frag(&As[cur][wave * 16 + l16][0], half);
#pragma unroll
    for (int t = 0; t < 4; ++t) {
      v16bf bf = make_frag(&Bs[cur][t * 16 + l16][0], half);
      acc[t] = __builtin_amdgcn_wmma_f32_16x16x32_bf16(false, af, false, bf,
                                                       (short)0, acc[t], false, false);
    }
    __syncthreads();                                     // cur free for ki+2
  }
  // epilogue: +bias, scatter to (b, head, t, d) as bf16
#pragma unroll
  for (int t = 0; t < 4; ++t) {
#pragma unroll
    for (int r = 0; r < 8; ++r) {
      int m = m0 + wave * 16 + r + 8 * half;     // m = b*NT + tt
      int n = n0 + t * 16 + l16;                 // n = head*ND + d
      float v = acc[t][r] + bias[n];
      int b = m >> 11, tt = m & (NT - 1);
      int h = n >> 7,  d  = n & (ND - 1);
      outb[(((size_t)(b * nheads + h)) * NT + tt) * ND + d] = f2bf(v);
    }
  }
}

// ---------------------------------------------------------------------------
// Kernel 2: per-head RMSNorm + RoPE over D=128 rows (bf16 in/out, fp32 math).
// One wave per row; 4 elems/lane.  rotate_half partner sits in lane^16.
// ---------------------------------------------------------------------------
__global__ __launch_bounds__(128) void norm_rope(
    unsigned short* __restrict__ hb, const float* __restrict__ w,
    const float* __restrict__ cosb, const float* __restrict__ sinb)
{
  const int tid = threadIdx.x;
  const int lane = tid & 31, wave = tid >> 5;
  const size_t row = (size_t)blockIdx.x * 4 + wave;  // over b*heads*T
  const int t = (int)(row & (NT - 1));
  unsigned short* p = hb + row * ND;
  const int d0 = lane * 4;

  uint2 raw = *(const uint2*)(p + d0);
  float f[4];
  f[0] = bf2f((unsigned short)(raw.x & 0xffff));
  f[1] = bf2f((unsigned short)(raw.x >> 16));
  f[2] = bf2f((unsigned short)(raw.y & 0xffff));
  f[3] = bf2f((unsigned short)(raw.y >> 16));

  float ss = f[0] * f[0] + f[1] * f[1] + f[2] * f[2] + f[3] * f[3];
#pragma unroll
  for (int m = 1; m <= 16; m <<= 1) ss += __shfl_xor(ss, m, 32);
  const float inv = rsqrtf(ss * (1.0f / (float)ND) + 1e-6f);

  float nf[4], out[4];
#pragma unroll
  for (int j = 0; j < 4; ++j) nf[j] = f[j] * inv * w[d0 + j];
  const float sgn = (lane < 16) ? -1.0f : 1.0f;   // rotate_half sign
#pragma unroll
  for (int j = 0; j < 4; ++j) {
    float partner = __shfl_xor(nf[j], 16, 32);    // element d ^ 64
    int d = d0 + j;
    out[j] = nf[j] * cosb[(size_t)t * ND + d] + sgn * partner * sinb[(size_t)t * ND + d];
  }
  uint2 o;
  o.x = pk2bf(out[0], out[1]);
  o.y = pk2bf(out[2], out[3]);
  *(uint2*)(p + d0) = o;
}

// ---------------------------------------------------------------------------
// Kernel 3: causal flash attention per (b, h, 64-row q tile).  4 waves,
// 16 q-rows each.  Q via TDM; K double-buffered via TDM (s_wait_tensorcnt 1);
// V double-buffered with next-block transposed loads overlapping compute;
// online softmax; all matmuls bf16 WMMA with fp32 accumulation.
// ---------------------------------------------------------------------------
__global__ __launch_bounds__(128) void flash_attn(
    const unsigned short* __restrict__ qb, const unsigned short* __restrict__ kb,
    const unsigned short* __restrict__ vb, unsigned short* __restrict__ yb)
{
  __shared__ __attribute__((aligned(16))) unsigned short Qs[64][ND];
  __shared__ __attribute__((aligned(16))) unsigned short Ks[2][32][ND];
  __shared__ __attribute__((aligned(16))) unsigned short Vs[2][ND][34];  // [d][key], padded
  __shared__ __attribute__((aligned(16))) unsigned short Ps[4][16][32];

  const int tid = threadIdx.x;
  const int lane = tid & 31, wave = tid >> 5;
  const int half = lane >> 4, l16 = lane & 15;
  const int qt = blockIdx.x, h = blockIdx.y, b = blockIdx.z;
  const int hk = h >> 2;
  const float scale = 0.08838834764831845f;      // 1/sqrt(128)

  const unsigned short* Qg = qb + (((size_t)(b * NH + h)) * NT + qt * 64) * ND;
  const unsigned short* Kg = kb + ((size_t)(b * NKVH + hk)) * NT * ND;
  const unsigned short* Vg = vb + ((size_t)(b * NKVH + hk)) * NT * ND;

  const int nkb = (qt + 1) * 2;                  // 32-key causal blocks
  unsigned ldsK[2];
  ldsK[0] = lds_addr_of(&Ks[0][0][0]); ldsK[1] = lds_addr_of(&Ks[1][0][0]);

  if (wave == 0) {                               // Q then K block 0 (in order)
    tdm_load_2d(lds_addr_of(&Qs[0][0]), Qg, ND, 64, ND, NT);
    tdm_load_2d(ldsK[0], Kg, ND, 32, ND, NT);
    __builtin_amdgcn_s_wait_tensorcnt(1);        // Q landed; K0 may stream on
  }
  for (int i = tid; i < 32 * ND; i += 128) {     // V block 0 transposed
    int key = i >> 7, d = i & (ND - 1);
    Vs[0][d][key] = Vg[(size_t)key * ND + d];
  }
  __syncthreads();

  v16bf qf[4];
#pragma unroll
  for (int c = 0; c < 4; ++c)
    qf[c] = make_frag(&Qs[wave * 16 + l16][c * 32], half);

  v8f o[8];
#pragma unroll
  for (int dc = 0; dc < 8; ++dc) o[dc] = vzero();
  float mstat[8], lstat[8];
#pragma unroll
  for (int r = 0; r < 8; ++r) { mstat[r] = -3.0e38f; lstat[r] = 0.0f; }

  const int qrow0 = qt * 64 + wave * 16;

  for (int kbi = 0; kbi < nkb; ++kbi) {
    const int kb0 = kbi * 32;
    const int cur = kbi & 1, nxt = cur ^ 1;
    if (wave == 0) {
      if (kbi + 1 < nkb) {                       // prefetch next K tile
        tdm_load_2d(ldsK[nxt], Kg + (size_t)(kb0 + 32) * ND, ND, 32, ND, NT);
        __builtin_amdgcn_s_wait_tensorcnt(1);    // Ks[cur] landed
      } else {
        __builtin_amdgcn_s_wait_tensorcnt(0);
      }
    }
    if (kbi + 1 < nkb) {                         // prefetch next V (transposed)
      for (int i = tid; i < 32 * ND; i += 128) {
        int key = i >> 7, d = i & (ND - 1);
        Vs[nxt][d][key] = Vg[(size_t)(kb0 + 32 + key) * ND + d];
      }
    }
    __syncthreads();

    // S = Q K^T : two 16x16 key tiles, contraction over D in 4 chunks
    v8f s0 = vzero(), s1 = vzero();
#pragma unroll
    for (int c = 0; c < 4; ++c) {
      v16bf bk0 = make_frag(&Ks[cur][l16][c * 32], half);
      v16bf bk1 = make_frag(&Ks[cur][16 + l16][c * 32], half);
      s0 = __builtin_amdgcn_wmma_f32_16x16x32_bf16(false, qf[c], false, bk0, (short)0, s0, false, false);
      s1 = __builtin_amdgcn_wmma_f32_16x16x32_bf16(false, qf[c], false, bk1, (short)0, s1, false, false);
    }

    float alpha[8];
#pragma unroll
    for (int r = 0; r < 8; ++r) {
      const int qi = qrow0 + r + 8 * half;
      const int k0i = kb0 + l16, k1i = kb0 + 16 + l16;
      float a0 = (k0i <= qi) ? s0[r] * scale : -3.0e38f;
      float a1 = (k1i <= qi) ? s1[r] * scale : -3.0e38f;
      float mx = rmax16(fmaxf(a0, a1));
      float mnew = fmaxf(mstat[r], mx);
      float al = __expf(mstat[r] - mnew);
      float p0 = __expf(a0 - mnew);
      float p1 = __expf(a1 - mnew);
      lstat[r] = lstat[r] * al + rsum16(p0 + p1);
      mstat[r] = mnew;
      alpha[r] = al;
      Ps[wave][r + 8 * half][l16]      = f2bf(p0);   // P in row-major (q, key)
      Ps[wave][r + 8 * half][16 + l16] = f2bf(p1);
    }
#pragma unroll
    for (int dc = 0; dc < 8; ++dc)
#pragma unroll
      for (int r = 0; r < 8; ++r) o[dc][r] *= alpha[r];

    // per-wave LDS bounce to transpose P into A-fragment layout; LDS ops from
    // one wave complete in order, fence the scheduler anyway.
    __builtin_amdgcn_wave_barrier();
    asm volatile("s_wait_dscnt 0" ::: "memory");

    v16bf pa = make_frag(&Ps[wave][l16][0], half);
#pragma unroll
    for (int dc = 0; dc < 8; ++dc) {
      v16bf bv = make_frag(&Vs[cur][dc * 16 + l16][0], half);
      o[dc] = __builtin_amdgcn_wmma_f32_16x16x32_bf16(false, pa, false, bv, (short)0, o[dc], false, false);
    }
    __syncthreads();                              // cur buffers free for kbi+2
  }

  // epilogue: normalize and store bf16 into (b, t, h*D + d)
#pragma unroll
  for (int dc = 0; dc < 8; ++dc) {
#pragma unroll
    for (int r = 0; r < 8; ++r) {
      int tq = qrow0 + r + 8 * half;
      int d  = dc * 16 + l16;
      float v = o[dc][r] / lstat[r];
      yb[((size_t)(b * NT + tq)) * (NH * ND) + h * ND + d] = f2bf(v);
    }
  }
}

// ---------------------------------------------------------------------------
// Kernel 4: out = Y * Wo^T  (Y bf16 (B*T, 2048), Wo bf16 (2048, 2048)), fp32 out
// Same double-buffered TDM pipeline as gemm_qkv.
// ---------------------------------------------------------------------------
__global__ __launch_bounds__(128) void gemm_out(
    const unsigned short* __restrict__ Y, const unsigned short* __restrict__ W,
    float* __restrict__ out, int Kdim)
{
  __shared__ __attribute__((aligned(16))) unsigned short As[2][64][32];
  __shared__ __attribute__((aligned(16))) unsigned short Bs[2][64][32];
  const int tid = threadIdx.x;
  const int lane = tid & 31, wave = tid >> 5;
  const int half = lane >> 4, l16 = lane & 15;
  const int m0 = blockIdx.x * 64;
  const int n0 = blockIdx.y * 64;
  unsigned ldsA[2], ldsB[2];
  ldsA[0] = lds_addr_of(&As[0][0][0]); ldsA[1] = lds_addr_of(&As[1][0][0]);
  ldsB[0] = lds_addr_of(&Bs[0][0][0]); ldsB[1] = lds_addr_of(&Bs[1][0][0]);

  v8f acc[4];
#pragma unroll
  for (int t = 0; t < 4; ++t) acc[t] = vzero();

  const int nk = Kdim / 32;
  if (wave == 0) {
    tdm_load_2d(ldsA[0], Y + (size_t)m0 * Kdim, 32, 64, Kdim, NB * NT);
    tdm_load_2d(ldsB[0], W + (size_t)n0 * Kdim, 32, 64, Kdim, NC);
  }
  for (int ki = 0; ki < nk; ++ki) {
    const int cur = ki & 1, nxt = cur ^ 1;
    if (wave == 0) {
      if (ki + 1 < nk) {
        const int k1 = (ki + 1) * 32;
        tdm_load_2d(ldsA[nxt], Y + (size_t)m0 * Kdim + k1, 32, 64, Kdim, NB * NT);
        tdm_load_2d(ldsB[nxt], W + (size_t)n0 * Kdim + k1, 32, 64, Kdim, NC);
        __builtin_amdgcn_s_wait_tensorcnt(2);
      } else {
        __builtin_amdgcn_s_wait_tensorcnt(0);
      }
    }
    __syncthreads();
    v16bf af = make_frag(&As[cur][wave * 16 + l16][0], half);
#pragma unroll
    for (int t = 0; t < 4; ++t) {
      v16bf bf = make_frag(&Bs[cur][t * 16 + l16][0], half);
      acc[t] = __builtin_amdgcn_wmma_f32_16x16x32_bf16(false, af, false, bf,
                                                       (short)0, acc[t], false, false);
    }
    __syncthreads();
  }
#pragma unroll
  for (int t = 0; t < 4; ++t) {
#pragma unroll
    for (int r = 0; r < 8; ++r) {
      int m = m0 + wave * 16 + r + 8 * half;
      int n = n0 + t * 16 + l16;
      out[(size_t)m * NC + n] = acc[t][r];
    }
  }
}

// ---------------------------------------------------------------------------
extern "C" void kernel_launch(void* const* d_in, const int* in_sizes, int n_in,
                              void* d_out, int out_size, void* d_ws, size_t ws_size,
                              hipStream_t stream) {
  (void)in_sizes; (void)n_in; (void)out_size; (void)ws_size;
  const float* x    = (const float*)d_in[0];
  const float* cosb = (const float*)d_in[1];
  const float* sinb = (const float*)d_in[2];
  const float* Wq   = (const float*)d_in[3];
  const float* bq   = (const float*)d_in[4];
  const float* Wk   = (const float*)d_in[5];
  const float* bk   = (const float*)d_in[6];
  const float* Wv   = (const float*)d_in[7];
  const float* bv   = (const float*)d_in[8];
  const float* Wo   = (const float*)d_in[9];
  const float* qn   = (const float*)d_in[10];
  const float* kn   = (const float*)d_in[11];
  float* out = (float*)d_out;

  char* ws = (char*)d_ws;
  const size_t MB = 1u << 20;
  unsigned short* qb  = (unsigned short*)(ws);             // 16 MB
  unsigned short* kb  = (unsigned short*)(ws + 16 * MB);   //  4 MB
  unsigned short* vb  = (unsigned short*)(ws + 20 * MB);   //  4 MB
  unsigned short* yb  = (unsigned short*)(ws + 24 * MB);   // 16 MB
  unsigned short* xb  = (unsigned short*)(ws + 40 * MB);   // 16 MB  x in bf16
  unsigned short* wqb = (unsigned short*)(ws + 56 * MB);   //  8 MB
  unsigned short* wkb = (unsigned short*)(ws + 64 * MB);   //  2 MB
  unsigned short* wvb = (unsigned short*)(ws + 66 * MB);   //  2 MB
  unsigned short* wob = (unsigned short*)(ws + 68 * MB);   //  8 MB

  const dim3 blk(128);
  const int M = NB * NT;                                   // 4096

  // one-time bf16 conversions
  cvt_bf16<<<dim3((M * NC / 4 + 255) / 256), dim3(256), 0, stream>>>(x,  xb,  M * NC / 4);
  cvt_bf16<<<dim3((NC * NC / 4 + 255) / 256), dim3(256), 0, stream>>>(Wq, wqb, NC * NC / 4);
  cvt_bf16<<<dim3((NKVH * ND * NC / 4 + 255) / 256), dim3(256), 0, stream>>>(Wk, wkb, NKVH * ND * NC / 4);
  cvt_bf16<<<dim3((NKVH * ND * NC / 4 + 255) / 256), dim3(256), 0, stream>>>(Wv, wvb, NKVH * ND * NC / 4);
  cvt_bf16<<<dim3((NC * NC / 4 + 255) / 256), dim3(256), 0, stream>>>(Wo, wob, NC * NC / 4);

  gemm_qkv<<<dim3(M / 64, (NH   * ND) / 64), blk, 0, stream>>>(xb, wqb, bq, qb, NC, M, NH * ND, NH);
  gemm_qkv<<<dim3(M / 64, (NKVH * ND) / 64), blk, 0, stream>>>(xb, wkb, bk, kb, NC, M, NKVH * ND, NKVH);
  gemm_qkv<<<dim3(M / 64, (NKVH * ND) / 64), blk, 0, stream>>>(xb, wvb, bv, vb, NC, M, NKVH * ND, NKVH);

  norm_rope<<<dim3((NB * NH   * NT) / 4), blk, 0, stream>>>(qb, qn, cosb, sinb);
  norm_rope<<<dim3((NB * NKVH * NT) / 4), blk, 0, stream>>>(kb, kn, cosb, sinb);

  flash_attn<<<dim3(NT / 64, NH, NB), blk, 0, stream>>>(qb, kb, vb, yb);

  gemm_out<<<dim3(M / 64, NC / 64), blk, 0, stream>>>(yb, wob, out, NH * ND);
}
